// GCN_PYG_58110907515588
// MI455X (gfx1250) — matrix operand order repaired and checked by reading, not compiled, not run on hardware
//
#include <hip/hip_runtime.h>

// ---------------------------------------------------------------------------
// GCN 2-layer forward for MI455X (gfx1250, wave32).
//   h  = relu( Dhat A Dhat (X @ W1) + b1 )
//   y  = log_softmax( Dhat A Dhat (h @ W2) + b2 )
// GEMMs use v_wmma_f32_16x16x32_bf16 (fp32 accum). x tiles staged to LDS with
// GLOBAL_LOAD_ASYNC_TO_LDS_B128. B (weight) fragments are precomputed per
// (k-step, wave, lane) into bf16 tables in the exact WMMA register image, so
// the inner loop loads them with two global_load_b128 (L2-resident) instead
// of 16 serialized strided scalar loads + conversion. Edge phases are
// wave-per-edge coalesced gather + f32 L2 atomics scatter.
// ---------------------------------------------------------------------------

typedef __bf16 v16bf __attribute__((ext_vector_type(16)));
typedef float  v8f   __attribute__((ext_vector_type(8)));
typedef unsigned int v4u __attribute__((ext_vector_type(4)));

union BFrag {
    v16bf v;
    unsigned short u[16];
    unsigned int   w[8];
    v4u            q[2];
    __device__ BFrag() {}
};

// pack two f32 -> bf16x2 (round-half-up), one v_perm_b32 for the packing:
// result = (bf16(hi) << 16) | bf16(lo)
__device__ __forceinline__ unsigned int pack_bf16x2(float lo, float hi) {
    unsigned int a = __float_as_uint(hi) + 0x8000u;
    unsigned int b = __float_as_uint(lo) + 0x8000u;
    // bytes: [0]=b[2] [1]=b[3] [2]=a[2] [3]=a[3]
    return __builtin_amdgcn_perm(a, b, 0x07060302u);
}

#if defined(__gfx1250__) && __has_builtin(__builtin_amdgcn_global_load_async_to_lds_b128)
#define HAVE_ASYNC_LDS 1
// exact parameter types per hipcc diagnostic: pointer-to-v4i in AS1 / AS3
typedef int v4i __attribute__((vector_size(16)));
typedef v4i __attribute__((address_space(1))) as1_v4i;
typedef v4i __attribute__((address_space(3))) as3_v4i;
#else
#define HAVE_ASYNC_LDS 0
#endif

__device__ __forceinline__ void wait_async0() {
#if __has_builtin(__builtin_amdgcn_s_wait_asynccnt)
    __builtin_amdgcn_s_wait_asynccnt(0);
#else
    asm volatile("s_wait_asynccnt 0x0" ::: "memory");
#endif
}

// ---------------------------------------------------------------------------
// init helpers
// ---------------------------------------------------------------------------
__global__ void fill_f32(float* __restrict__ p, float v, long long n) {
    long long i = (long long)blockIdx.x * blockDim.x + threadIdx.x;
    if (i < n) p[i] = v;
}

// out[i*40 + j] = b2[j]  (bias-seeded accumulator for layer-2 aggregation)
__global__ void init_bias40(float* __restrict__ out, const float* __restrict__ b2,
                            long long n40) {
    long long i = (long long)blockIdx.x * blockDim.x + threadIdx.x;
    if (i < n40) out[i] = b2[(int)(i % 40)];
}

// ---------------------------------------------------------------------------
// degree: deg = 1 (self loop) + sum ew over dst ; then dinv = rsqrt(deg)
// ---------------------------------------------------------------------------
__global__ void deg_accumulate(const long long* __restrict__ dst,
                               const float* __restrict__ ew,
                               float* __restrict__ deg, long long E) {
    long long e = (long long)blockIdx.x * blockDim.x + threadIdx.x;
    if (e < E) atomicAdd(deg + dst[e], ew[e]);
}

__global__ void deg_rsqrt(float* __restrict__ deg, int n) {
    int i = blockIdx.x * blockDim.x + threadIdx.x;
    if (i < n) {
        float d = deg[i];
        deg[i] = (d > 0.0f) ? rsqrtf(d) : 0.0f;
    }
}

// ---------------------------------------------------------------------------
// B-fragment tables: bf16 WMMA register image of W1 / W2.
//   w1tab[((ks*4 + wave)*32 + lane)*8 + d] ;  ks=0..15, wave=0..3
//   w2tab[((ks*3 + wave)*32 + lane)*8 + d] ;  ks=0..1,  wave=0..2 (n>=40 -> 0)
// B layout: lane half h, element j -> K offset 8h+j (j<8) or 16+8h+(j-8) (j>=8),
// n = wave*16 + (lane&15); dword d packs (j=2d, j=2d+1).
// ---------------------------------------------------------------------------
__global__ void build_w1tab(const float* __restrict__ W1, unsigned int* __restrict__ tab) {
    const int i = blockIdx.x * 256 + threadIdx.x;
    if (i >= 16 * 4 * 32 * 8) return;
    const int d    = i & 7;
    const int lane = (i >> 3) & 31;
    const int wave = (i >> 8) & 3;
    const int ks   = i >> 10;
    const int kb   = ks * 32 + (lane >> 4) * 8 + (d >> 2) * 16 + (d & 3) * 2;
    const int n    = wave * 16 + (lane & 15);
    tab[i] = pack_bf16x2(W1[(size_t)kb * 64 + n], W1[(size_t)(kb + 1) * 64 + n]);
}

__global__ void build_w2tab(const float* __restrict__ W2, unsigned int* __restrict__ tab) {
    const int i = blockIdx.x * 256 + threadIdx.x;
    if (i >= 2 * 3 * 32 * 8) return;
    const int d    = i & 7;
    const int lane = (i >> 3) & 31;
    const int g    = i >> 8;         // 0..5
    const int wave = g % 3;
    const int ks   = g / 3;
    const int kb   = ks * 32 + (lane >> 4) * 8 + (d >> 2) * 16 + (d & 3) * 2;
    const int n    = wave * 16 + (lane & 15);
    const float w0 = (n < 40) ? W2[kb * 40 + n] : 0.0f;
    const float w1 = (n < 40) ? W2[(kb + 1) * 40 + n] : 0.0f;
    tab[i] = pack_bf16x2(w0, w1);
}

// ---------------------------------------------------------------------------
// GEMM1: xw[N,64] = X[N,512] @ W1[512,64]   (bf16 WMMA, f32 accum)
// block: 128 threads = 4 waves; block owns 64 rows (4 M-tiles), wave w owns
// cols [16w,16w+16). One B fragment feeds 4 WMMAs -> 4x less W1 L2 traffic.
// ---------------------------------------------------------------------------
#define MT  4      // M tiles per block (64 rows)
#define TAW 36     // LDS row stride in floats: 144B (16B aligned, conflict-free)

__global__ void __launch_bounds__(128)
gemm_x_w1(const float* __restrict__ x, const unsigned int* __restrict__ w1tab,
          float* __restrict__ xw, int nrows) {
    __shared__ float tA[64 * TAW];           // 64 rows x 32 K of x (padded)
    const int lane = threadIdx.x & 31;
    const int wave = threadIdx.x >> 5;       // N-tile 0..3
    const int half = lane >> 4;              // lane group 0/1
    const int l16  = lane & 15;
    const int rowBase = blockIdx.x * (MT * 16);

    v8f acc[MT];
#pragma unroll
    for (int t = 0; t < MT; ++t)
        acc[t] = (v8f){0.f, 0.f, 0.f, 0.f, 0.f, 0.f, 0.f, 0.f};

    for (int k0 = 0; k0 < 512; k0 += 32) {
        __syncthreads();                     // guard tA reuse
        // stage 64x32 fp32 tile: 512 x 16B chunks, 4 per thread, coalesced
#pragma unroll
        for (int c = threadIdx.x; c < 512; c += 128) {
            const int r  = c >> 3;           // row in tile
            const int cw = (c & 7) << 2;     // float offset in row
            int row = rowBase + r;
            if (row >= nrows) row = nrows - 1;
            const float* gp = x + (size_t)row * 512 + k0 + cw;
            float* lp = tA + r * TAW + cw;
#if HAVE_ASYNC_LDS
            __builtin_amdgcn_global_load_async_to_lds_b128(
                (as1_v4i*)gp, (as3_v4i*)lp, 0, 0);
#else
            *(float4*)lp = *(const float4*)gp;
#endif
        }
        // B fragment: precomputed WMMA register image, two b128 loads
        BFrag b;
        {
            const unsigned int* bt =
                w1tab + ((size_t)((k0 >> 5) * 4 + wave) * 32 + lane) * 8;
            b.q[0] = *(const v4u*)bt;
            b.q[1] = *(const v4u*)(bt + 4);
        }
#if HAVE_ASYNC_LDS
        wait_async0();
#endif
        __syncthreads();

#pragma unroll
        for (int t = 0; t < MT; ++t) {
            // A 16x32 bf16: lane half -> K in {0..7,16..23} or {8..15,24..31}
            BFrag a;
            const float* ar = tA + (t * 16 + l16) * TAW + half * 8;
#pragma unroll
            for (int j = 0; j < 16; j += 2) {
                const int k = ((j >> 3) << 4) + (j & 7);
                a.w[j >> 1] = pack_bf16x2(ar[k], ar[k + 1]);
            }
            acc[t] = __builtin_amdgcn_wmma_f32_16x16x32_bf16(
                false, a.v, false, b.v, (short)0, acc[t], false, false);
        }
    }
    // C/D layout: vgpr r, M = half*8 + r, N = l16
#pragma unroll
    for (int t = 0; t < MT; ++t) {
        const int mBase = rowBase + t * 16 + half * 8;
        float* op = xw + (size_t)mBase * 64 + (wave << 4) + l16;
        if (rowBase + t * 16 + 16 <= nrows) {      // full-tile fast path
#pragma unroll
            for (int r = 0; r < 8; ++r) op[(size_t)r * 64] = acc[t][r];
        } else {
#pragma unroll
            for (int r = 0; r < 8; ++r)
                if (mBase + r < nrows) op[(size_t)r * 64] = acc[t][r];
        }
    }
}

// ---------------------------------------------------------------------------
// GEMM2: hw[N,40] = H[N,64] @ W2[64,40]   (N padded to 48 in-register)
// block: 96 threads = 3 waves (N tiles at 0,16,32)
// ---------------------------------------------------------------------------
__global__ void __launch_bounds__(96)
gemm_h_w2(const float* __restrict__ h, const unsigned int* __restrict__ w2tab,
          float* __restrict__ hw, int nrows) {
    const int lane = threadIdx.x & 31;
    const int wave = threadIdx.x >> 5;       // 0..2
    const int half = lane >> 4;
    const int l16  = lane & 15;
    const int rowBase = blockIdx.x * 16;
    int arow = rowBase + l16;
    if (arow >= nrows) arow = nrows - 1;
    const int n = (wave << 4) + l16;         // 0..47

    v8f acc = {0.f, 0.f, 0.f, 0.f, 0.f, 0.f, 0.f, 0.f};
#pragma unroll
    for (int k0 = 0; k0 < 64; k0 += 32) {
        BFrag a, b;
        {
            const unsigned int* bt =
                w2tab + ((size_t)((k0 >> 5) * 3 + wave) * 32 + lane) * 8;
            b.q[0] = *(const v4u*)bt;
            b.q[1] = *(const v4u*)(bt + 4);
        }
        const float* ap = h + (size_t)arow * 64 + k0 + half * 8;
#pragma unroll
        for (int j = 0; j < 16; j += 2) {
            const int k = ((j >> 3) << 4) + (j & 7);
            a.w[j >> 1] = pack_bf16x2(ap[k], ap[k + 1]);
        }
        acc = __builtin_amdgcn_wmma_f32_16x16x32_bf16(
            false, a.v, false, b.v, (short)0, acc, false, false);
    }
    if (n < 40) {
        float* op = hw + (size_t)(rowBase + half * 8) * 40 + n;
        if (rowBase + 16 <= nrows) {               // full-tile fast path
#pragma unroll
            for (int r = 0; r < 8; ++r) op[(size_t)r * 40] = acc[r];
        } else {
#pragma unroll
            for (int r = 0; r < 8; ++r)
                if (rowBase + half * 8 + r < nrows) op[(size_t)r * 40] = acc[r];
        }
    }
}

// ---------------------------------------------------------------------------
// edge aggregation, 64-wide features: out[dst] += xw[src] * norm
// one wave per edge; lanes cover the contiguous 256B feature row.
// grid-stride with global_prefetch of the next gather row.
// ---------------------------------------------------------------------------
__global__ void __launch_bounds__(256)
aggregate64(const float* __restrict__ xw, const long long* __restrict__ src,
            const long long* __restrict__ dst, const float* __restrict__ ew,
            const float* __restrict__ dinv, float* __restrict__ out,
            long long E, long long total) {
    const int lane = threadIdx.x & 31;
    const long long stride = (long long)gridDim.x * (blockDim.x >> 5);
    long long e = (long long)blockIdx.x * (blockDim.x >> 5) + (threadIdx.x >> 5);
    for (; e < total; e += stride) {
        long long s, d; float w;
        if (e < E) { s = src[e]; d = dst[e]; w = ew[e]; }
        else       { s = e - E; d = s;       w = 1.0f; }
        const long long en = e + stride;
        if (en < E) __builtin_prefetch(xw + src[en] * 64 + lane, 0, 0);
        const float norm = dinv[s] * w * dinv[d];
        const float* m = xw + s * 64;
        float* o = out + d * 64;
        atomicAdd(o + lane,      m[lane]      * norm);
        atomicAdd(o + lane + 32, m[lane + 32] * norm);
    }
}

// 40-wide variant for layer 2
__global__ void __launch_bounds__(256)
aggregate40(const float* __restrict__ hw, const long long* __restrict__ src,
            const long long* __restrict__ dst, const float* __restrict__ ew,
            const float* __restrict__ dinv, float* __restrict__ out,
            long long E, long long total) {
    const int lane = threadIdx.x & 31;
    const long long stride = (long long)gridDim.x * (blockDim.x >> 5);
    long long e = (long long)blockIdx.x * (blockDim.x >> 5) + (threadIdx.x >> 5);
    for (; e < total; e += stride) {
        long long s, d; float w;
        if (e < E) { s = src[e]; d = dst[e]; w = ew[e]; }
        else       { s = e - E; d = s;       w = 1.0f; }
        const long long en = e + stride;
        if (en < E) __builtin_prefetch(hw + src[en] * 40 + lane, 0, 0);
        const float norm = dinv[s] * w * dinv[d];
        const float* m = hw + s * 40;
        float* o = out + d * 40;
        atomicAdd(o + lane, m[lane] * norm);
        if (lane < 8) atomicAdd(o + lane + 32, m[lane + 32] * norm);
    }
}

// h = relu(agg + b1), elementwise over N*64 (in place)
__global__ void relu_bias64(float* __restrict__ h, const float* __restrict__ b1,
                            long long n64) {
    long long i = (long long)blockIdx.x * blockDim.x + threadIdx.x;
    if (i < n64) h[i] = fmaxf(h[i] + b1[(int)(i & 63)], 0.0f);
}

// in-place row-wise log_softmax over 40 classes; one wave per row
__global__ void __launch_bounds__(256)
log_softmax40(float* __restrict__ out, int nrows) {
    const int lane = threadIdx.x & 31;
    const int row = blockIdx.x * 8 + (threadIdx.x >> 5);
    if (row >= nrows) return;
    float* p = out + (size_t)row * 40;
    const float v0 = p[lane];                          // cols 0..31
    const float v1 = (lane < 8) ? p[lane + 32] : -3.0e38f;
    float m = fmaxf(v0, v1);
#pragma unroll
    for (int off = 16; off >= 1; off >>= 1)
        m = fmaxf(m, __shfl_xor(m, off, 32));
    float s = expf(v0 - m) + ((lane < 8) ? expf(v1 - m) : 0.0f);
#pragma unroll
    for (int off = 16; off >= 1; off >>= 1)
        s += __shfl_xor(s, off, 32);
    const float lse = logf(s) + m;
    p[lane] = v0 - lse;
    if (lane < 8) p[lane + 32] = v1 - lse;
}

// ---------------------------------------------------------------------------
// launcher
// ---------------------------------------------------------------------------
extern "C" void kernel_launch(void* const* d_in, const int* in_sizes, int n_in,
                              void* d_out, int out_size, void* d_ws, size_t ws_size,
                              hipStream_t stream) {
    const float*     x   = (const float*)d_in[0];      // [N,512]
    const long long* ei  = (const long long*)d_in[1];  // [2,E] int64
    const float*     ew  = (const float*)d_in[2];      // [E]
    const float*     W1  = (const float*)d_in[3];      // [512,64]
    const float*     b1  = (const float*)d_in[4];      // [64]
    const float*     W2  = (const float*)d_in[5];      // [64,40]
    const float*     b2  = (const float*)d_in[6];      // [40]
    float*           out = (float*)d_out;              // [N,40]

    const int       N     = in_sizes[0] / 512;
    const long long E     = in_sizes[2];
    const long long total = E + (long long)N;
    const long long* src  = ei;
    const long long* dst  = ei + E;

    // workspace layout (fp32 unless noted):
    //   dinv  : N
    //   xw    : N*64   (X@W1; later reused for H@W2 since only N*40 needed)
    //   h1    : N*64   (layer-1 aggregate -> relu'd hidden)
    //   w1tab : 16384 u32 (64KB),  w2tab : 1536 u32 (6KB)
    char* ws = (char*)d_ws;
    float* dinv = (float*)ws;
    size_t off  = (((size_t)N * 4) + 255) & ~(size_t)255;
    float* xw   = (float*)(ws + off);
    float* h1   = xw + (size_t)N * 64;
    unsigned int* w1tab = (unsigned int*)(h1 + (size_t)N * 64);
    unsigned int* w2tab = w1tab + 16 * 4 * 32 * 8;

    const long long n64 = (long long)N * 64;
    const long long n40 = (long long)N * 40;
    const int aggBlocks = 32768;             // grid-stride edge loop

    // 0) build bf16 B-fragment tables (tiny, L2-resident afterwards)
    build_w1tab<<<(16 * 4 * 32 * 8 + 255) / 256, 256, 0, stream>>>(W1, w1tab);
    build_w2tab<<<(2 * 3 * 32 * 8 + 255) / 256, 256, 0, stream>>>(W2, w2tab);

    // 1) dinv = rsqrt(1 + segment_sum(ew, dst))
    fill_f32<<<(N + 255) / 256, 256, 0, stream>>>(dinv, 1.0f, N);
    deg_accumulate<<<(int)((E + 255) / 256), 256, 0, stream>>>(dst, ew, dinv, E);
    deg_rsqrt<<<(N + 255) / 256, 256, 0, stream>>>(dinv, N);

    // 2) xw = X @ W1  (WMMA bf16, async LDS staging)
    gemm_x_w1<<<(N + 63) / 64, 128, 0, stream>>>(x, w1tab, xw, N);

    // 3) h1 = Dhat A Dhat xw ; h1 = relu(h1 + b1)
    fill_f32<<<(int)((n64 + 255) / 256), 256, 0, stream>>>(h1, 0.0f, n64);
    aggregate64<<<aggBlocks, 256, 0, stream>>>(xw, src, dst, ew, dinv, h1, E, total);
    relu_bias64<<<(int)((n64 + 255) / 256), 256, 0, stream>>>(h1, b1, n64);

    // 4) hw = h1 @ W2 (WMMA bf16), stored into xw buffer (reused)
    gemm_h_w2<<<(N + 15) / 16, 96, 0, stream>>>(h1, w2tab, xw, N);

    // 5) out = b2 + Dhat A Dhat hw ; log_softmax rows
    init_bias40<<<(int)((n40 + 255) / 256), 256, 0, stream>>>(out, b2, n40);
    aggregate40<<<aggBlocks, 256, 0, stream>>>(xw, src, dst, ew, dinv, out, E, total);
    log_softmax40<<<(N + 7) / 8, 256, 0, stream>>>(out, N);
}
